// PCmer_6640019439746
// MI455X (gfx1250) — compile-verified, matrix operand order checked by reading, not run on hardware
//
#include <hip/hip_runtime.h>
#include <hip/hip_bf16.h>
#include <math.h>

// ---------------------------------------------------------------------------
// Problem constants (from reference)
// ---------------------------------------------------------------------------
#define LL 4
#define BB 8
#define NN 2048
#define DD 512
#define HH 8
#define DH 64
#define MF 266          // int(64*ln(64))
#define INNER 1024      // 2*D
#define KCONV 31
#define KEPS 1.0e-4f

typedef __attribute__((ext_vector_type(16))) __bf16 v16bf;
typedef __attribute__((ext_vector_type(8)))  float  v8f;
typedef __attribute__((ext_vector_type(4)))  unsigned v4u;
typedef __attribute__((ext_vector_type(8)))  int    v8i;
typedef __attribute__((ext_vector_type(4)))  int    v4i;

#define BM 128
#define BN 128
#define BK 32
#define LDP (BK + 8)     // LDS row stride in bf16 (80B, 16B aligned)

// ---------------------------------------------------------------------------
// Tensor Data Mover: load a fully-in-bounds [rows x BK] bf16 tile (row stride
// `ld` elements in global) into LDS at `lds_off`, with LDS padding of 4 DWORDs
// after every 16 DWORDs (reproduces the [rows][BK+8] layout).
// D# packing per CDNA5 ISA.
// This toolchain exposes the 6-arg builtin:
//   (uint32x4 g0, int32x8 g1, int32x4 g2, int32x4 g3, int32x8 gx, i32 cpol)
// G2/G3 are NULL (zero) for <=2D tensors.
// ---------------------------------------------------------------------------
#if __has_builtin(__builtin_amdgcn_tensor_load_to_lds)
#define HAVE_TDM 1
__device__ __forceinline__ void tdm_load_tile(const __bf16* gsrc, int ld,
                                              unsigned lds_off, int rows)
{
    unsigned long long ga = (unsigned long long)(const void*)gsrc;
    v4u g0;
    g0[0] = 1u;                                              // count=1 (valid D#)
    g0[1] = lds_off;                                         // lds_addr [63:32]
    g0[2] = (unsigned)(ga & 0xffffffffu);                    // global_addr lo
    g0[3] = (unsigned)((ga >> 32) & 0x1ffffffu) | (2u << 30);// addr hi + type=2

    const unsigned d0 = (unsigned)BK;       // tile/tensor dim0 (contiguous, elems)
    const unsigned d1 = (unsigned)rows;     // tile/tensor dim1
    const unsigned long long s0 = (unsigned long long)(unsigned)ld;

    v8i g1;
    g1[0] = (int)((1u << 16)    // data_size = 2B
                | (1u << 20)    // pad_enable
                | (3u << 22)    // pad_interval: 16 DWORDs
                | (3u << 25));  // pad_amount:   4 DWORDs (=8 bf16)
    g1[1] = (int)((d0 & 0xffffu) << 16);                     // tensor_dim0[15:0]
    g1[2] = (int)((d0 >> 16) | ((d1 & 0xffffu) << 16));      // dim0 hi | dim1 lo
    g1[3] = (int)((d1 >> 16) | ((unsigned)BK << 16));        // dim1 hi | tile_dim0
    g1[4] = (int)(d1 & 0xffffu);                             // tile_dim1 (tile_dim2=0)
    g1[5] = (int)(s0 & 0xffffffffu);                         // tensor_dim0_stride lo
    g1[6] = (int)((s0 >> 32) & 0xffffu);                     // stride hi | dim1_stride=0
    g1[7] = 0;
    v4i z4 = (v4i){0, 0, 0, 0};
    v8i z8 = (v8i){0, 0, 0, 0, 0, 0, 0, 0};
    __builtin_amdgcn_tensor_load_to_lds(g0, g1, z4, z4, z8, 0);
}
#else
#define HAVE_TDM 0
#endif

// ---------------------------------------------------------------------------
// Batched WMMA GEMM (bf16 in, fp32 accumulate):
//   C[z] = alpha * opA(A[z]) * opB(B[z])  (*rowscale) (+bias) (+resid)
// opA(m,k) = TA ? A[k*lda+m] : A[m*lda+k]
// opB(k,n) = TB ? B[n*ldb+k] : B[k*ldb+n]
// 128x128 block tile, 8 wave32 waves (2x4), wave = 4x2 WMMA 16x16x32 tiles.
// CB=1 -> store bf16, else fp32.
// ---------------------------------------------------------------------------
template <int TA, int TB, int CB>
__global__ __launch_bounds__(256)
void wmma_gemm(const __bf16* __restrict__ A, const __bf16* __restrict__ Bm,
               const float* __restrict__ bias, const float* __restrict__ resid,
               const float* __restrict__ rowscale, void* __restrict__ Cv,
               int Mr, int Nc, int Kd, int lda, int ldb, int ldc,
               long long sA, long long sB, long long sC, long long sScale,
               float alpha)
{
    const int z = blockIdx.z;
    A  += (size_t)z * (size_t)sA;
    Bm += (size_t)z * (size_t)sB;
    float*  Cf = (float*)Cv + (size_t)z * (size_t)sC;
    __bf16* Cb = (__bf16*)Cv + (size_t)z * (size_t)sC;
    if (rowscale) rowscale += (size_t)z * (size_t)sScale;

    __shared__ __bf16 As[BM][LDP];
    __shared__ __bf16 Bs[BN][LDP];

    const int tid  = threadIdx.x;
    const int lane = tid & 31;
    const int wid  = tid >> 5;
    const int waveM = wid >> 2;
    const int waveN = wid & 3;

    const int m0 = blockIdx.y * BM;
    const int n0 = blockIdx.x * BN;

    const bool fullM = (m0 + BM <= Mr);
    const bool fullN = (n0 + BN <= Nc);

    v8f acc[4][2];
#pragma unroll
    for (int mi = 0; mi < 4; ++mi)
#pragma unroll
        for (int ni = 0; ni < 2; ++ni)
            acc[mi][ni] = (v8f){0.f, 0.f, 0.f, 0.f, 0.f, 0.f, 0.f, 0.f};

    for (int kb = 0; kb < Kd; kb += BK) {
        const bool fullK = (kb + BK <= Kd);
#if HAVE_TDM
        const bool tdmA = (TA == 0) && fullM && fullK;
        const bool tdmB = (TB == 1) && fullN && fullK;
#else
        const bool tdmA = false, tdmB = false;
#endif
#if HAVE_TDM
        if (tdmA) {
            if (wid == 0)
                tdm_load_tile(A + (size_t)m0 * lda + kb, lda,
                              (unsigned)(size_t)&As[0][0], BM);
        } else
#endif
        {
            for (int idx = tid; idx < BM * BK; idx += 256) {
                const int m = idx >> 5, k = idx & 31;
                const int gm = m0 + m, gk = kb + k;
                __bf16 v = (__bf16)0.f;
                if (gm < Mr && gk < Kd)
                    v = TA ? A[(size_t)gk * lda + gm] : A[(size_t)gm * lda + gk];
                As[m][k] = v;
            }
        }
#if HAVE_TDM
        if (tdmB) {
            if (wid == 0)
                tdm_load_tile(Bm + (size_t)n0 * ldb + kb, ldb,
                              (unsigned)(size_t)&Bs[0][0], BN);
        } else
#endif
        {
            for (int idx = tid; idx < BN * BK; idx += 256) {
                const int n = idx >> 5, k = idx & 31;
                const int gn = n0 + n, gk = kb + k;
                __bf16 v = (__bf16)0.f;
                if (gn < Nc && gk < Kd)
                    v = TB ? Bm[(size_t)gn * ldb + gk] : Bm[(size_t)gk * ldb + gn];
                Bs[n][k] = v;
            }
        }
#if HAVE_TDM
        if ((tdmA || tdmB) && wid == 0)
            __builtin_amdgcn_s_wait_tensorcnt(0);
#endif
        __syncthreads();

        // fragment loads per ISA 16-bit A/B lane layouts
        const int kh = (lane >> 4) << 3;
        const int lr = lane & 15;

        v16bf afr[4], bfr[2];
#pragma unroll
        for (int mi = 0; mi < 4; ++mi) {
            const int r = waveM * 64 + mi * 16 + lr;
#pragma unroll
            for (int j = 0; j < 8; ++j) {
                afr[mi][j]     = As[r][kh + j];
                afr[mi][j + 8] = As[r][16 + kh + j];
            }
        }
#pragma unroll
        for (int ni = 0; ni < 2; ++ni) {
            const int cno = waveN * 32 + ni * 16 + lr;
#pragma unroll
            for (int j = 0; j < 8; ++j) {
                bfr[ni][j]     = Bs[cno][kh + j];
                bfr[ni][j + 8] = Bs[cno][16 + kh + j];
            }
        }
#pragma unroll
        for (int mi = 0; mi < 4; ++mi)
#pragma unroll
            for (int ni = 0; ni < 2; ++ni)
                acc[mi][ni] = __builtin_amdgcn_wmma_f32_16x16x32_bf16(
                    false, afr[mi], false, bfr[ni], (short)0, acc[mi][ni],
                    false, false);
        __syncthreads();
    }

    // epilogue: C/D layout  VGPR r -> row = r + 8*(lane>=16), col = lane&15
    const int rbase = (lane >> 4) * 8;
    const int cix   = lane & 15;
#pragma unroll
    for (int mi = 0; mi < 4; ++mi) {
#pragma unroll
        for (int ni = 0; ni < 2; ++ni) {
#pragma unroll
            for (int r = 0; r < 8; ++r) {
                const int grow = m0 + waveM * 64 + mi * 16 + rbase + r;
                const int gcol = n0 + waveN * 32 + ni * 16 + cix;
                if (grow < Mr && gcol < Nc) {
                    float v = alpha * acc[mi][ni][r];
                    if (rowscale) v *= rowscale[grow];
                    if (bias)     v += bias[gcol];
                    if (resid)    v += resid[(size_t)grow * ldc + gcol];
                    if (CB) Cb[(size_t)grow * ldc + gcol] = (__bf16)v;
                    else    Cf[(size_t)grow * ldc + gcol] = v;
                }
            }
        }
    }
}

// ---------------------------------------------------------------------------
// fp32 -> bf16 converter (weights)
// ---------------------------------------------------------------------------
__global__ void cvt_bf16_kernel(const float* __restrict__ s, __bf16* __restrict__ d,
                                long long n)
{
    long long i = (long long)blockIdx.x * 256 + threadIdx.x;
    if (i < n) d[i] = (__bf16)s[i];
}

// LayerNorm (fp32 in, bf16 out). One 256-thread block per row.
__global__ __launch_bounds__(256)
void ln_kernel(const float* __restrict__ x, const float* __restrict__ g,
               const float* __restrict__ b, __bf16* __restrict__ y, int Dn)
{
    const int row = blockIdx.x;
    const float* xr = x + (size_t)row * Dn;
    __bf16* yr = y + (size_t)row * Dn;
    __shared__ float red[8];
    const int tid = threadIdx.x, lane = tid & 31, wid = tid >> 5;

    float s = 0.f;
    for (int j = tid; j < Dn; j += 256) s += xr[j];
    for (int o = 16; o > 0; o >>= 1) s += __shfl_down(s, o, 32);
    if (lane == 0) red[wid] = s;
    __syncthreads();
    float mu = 0.f;
#pragma unroll
    for (int w = 0; w < 8; ++w) mu += red[w];
    mu /= (float)Dn;
    __syncthreads();

    float v = 0.f;
    for (int j = tid; j < Dn; j += 256) { float d = xr[j] - mu; v += d * d; }
    for (int o = 16; o > 0; o >>= 1) v += __shfl_down(v, o, 32);
    if (lane == 0) red[wid] = v;
    __syncthreads();
    float var = 0.f;
#pragma unroll
    for (int w = 0; w < 8; ++w) var += red[w];
    const float rstd = rsqrtf(var / (float)Dn + 1e-5f);

    for (int j = tid; j < Dn; j += 256)
        yr[j] = (__bf16)((xr[j] - mu) * rstd * g[j] + b[j]);
}

// [B,N,H,DH] -> [B,H,N,DH]  (bf16) and inverse
__global__ void split_heads_kernel(const __bf16* __restrict__ src, __bf16* __restrict__ dst)
{
    size_t i = (size_t)blockIdx.x * blockDim.x + threadIdx.x;
    if (i >= (size_t)BB * NN * HH * DH) return;
    const int d = (int)(i % DH); size_t t = i / DH;
    const int h = (int)(t % HH); t /= HH;
    const int n = (int)(t % NN); const int b = (int)(t / NN);
    dst[(((size_t)b * HH + h) * NN + n) * DH + d] = src[i];
}

__global__ void merge_heads_kernel(const __bf16* __restrict__ src, __bf16* __restrict__ dst)
{
    size_t i = (size_t)blockIdx.x * blockDim.x + threadIdx.x;
    if (i >= (size_t)BB * HH * NN * DH) return;
    const int d = (int)(i % DH); size_t t = i / DH;
    const int n = (int)(t % NN); t /= NN;
    const int h = (int)(t % HH); const int b = (int)(t / HH);
    dst[(((size_t)b * NN + n) * HH + h) * DH + d] = src[i];
}

// FAVOR+ epilogue: dd fp32 in, bf16 features out. block = 128 threads.
__global__ __launch_bounds__(128)
void feat_kernel(const float* __restrict__ dd, const __bf16* __restrict__ dat,
                 __bf16* __restrict__ ob, float nh, float ratio, int isq)
{
    const int row = blockIdx.x;
    const float* r = dd + (size_t)row * MF;
    __bf16* o = ob + (size_t)row * MF;
    const __bf16* dp = dat + (size_t)row * DH;
    __shared__ float red[4];
    const int tid = threadIdx.x, lane = tid & 31, wid = tid >> 5;

    float s = 0.f;
    for (int j = tid; j < DH; j += 128) { const float t = (float)dp[j]; s += t * t; }
    for (int o2 = 16; o2 > 0; o2 >>= 1) s += __shfl_down(s, o2, 32);
    if (lane == 0) red[wid] = s;
    __syncthreads();
    const float diag = (red[0] + red[1] + red[2] + red[3]) * nh;
    __syncthreads();

    if (isq) {
        float m = -3.4e38f;
        for (int j = tid; j < MF; j += 128) m = fmaxf(m, r[j]);
        for (int o2 = 16; o2 > 0; o2 >>= 1) m = fmaxf(m, __shfl_down(m, o2, 32));
        if (lane == 0) red[wid] = m;
        __syncthreads();
        const float mx = fmaxf(fmaxf(red[0], red[1]), fmaxf(red[2], red[3]));
        for (int j = tid; j < MF; j += 128)
            o[j] = (__bf16)(ratio * (expf(r[j] - diag - mx) + KEPS));
    } else {
        for (int j = tid; j < MF; j += 128)
            o[j] = (__bf16)(ratio * expf(r[j] - diag + KEPS));
    }
}

// ksum[bh, m] = sum_n kk[bh, n, m]
__global__ __launch_bounds__(256)
void ksum_kernel(const __bf16* __restrict__ kk, float* __restrict__ ks)
{
    const int bh = blockIdx.x;
    const __bf16* base = kk + (size_t)bh * NN * MF;
    for (int m = threadIdx.x; m < MF; m += 256) {
        float s = 0.f;
        for (int n = 0; n < NN; ++n) s += (float)base[(size_t)n * MF + m];
        ks[(size_t)bh * MF + m] = s;
    }
}

// d_inv[row] = 1 / dot(qk[row,:], ksum[bh,:]);  one wave per row
__global__ __launch_bounds__(256)
void dinv_kernel(const __bf16* __restrict__ qk, const float* __restrict__ ks,
                 float* __restrict__ dinv)
{
    const int wid = threadIdx.x >> 5, lane = threadIdx.x & 31;
    const size_t row = (size_t)blockIdx.x * 8 + wid;
    const size_t bh = row / NN;
    const __bf16* q = qk + row * MF;
    const float* k = ks + bh * MF;
    float s = 0.f;
    for (int j = lane; j < MF; j += 32) s += (float)q[j] * k[j];
    for (int o = 16; o > 0; o >>= 1) s += __shfl_down(s, o, 32);
    if (lane == 0) dinv[row] = 1.f / s;
}

// GLU (fp32 -> fp32)
__global__ void glu_kernel(const float* __restrict__ y, float* __restrict__ o)
{
    size_t i = (size_t)blockIdx.x * blockDim.x + threadIdx.x;
    if (i >= (size_t)BB * NN * INNER) return;
    const int c = (int)(i % INNER);
    const size_t row = i / INNER;
    const float a = y[row * (size_t)(2 * INNER) + c];
    const float g = y[row * (size_t)(2 * INNER) + INNER + c];
    o[i] = a * (1.f / (1.f + expf(-g)));
}

// depthwise conv (pad 15/15) + SiLU;  fp32 in, bf16 out;  layout [B, N, INNER]
__global__ void dwconv_silu_kernel(const float* __restrict__ y, const float* __restrict__ w,
                                   const float* __restrict__ bias, __bf16* __restrict__ z)
{
    size_t i = (size_t)blockIdx.x * blockDim.x + threadIdx.x;
    if (i >= (size_t)BB * NN * INNER) return;
    const int c = (int)(i % INNER); size_t t = i / INNER;
    const int n = (int)(t % NN); const int b = (int)(t / NN);
    const float* wr = w + (size_t)c * KCONV;
    float acc = bias[c];
#pragma unroll
    for (int k = 0; k < KCONV; ++k) {
        const int nn = n + k - (KCONV / 2);
        if (nn >= 0 && nn < NN)
            acc += y[((size_t)b * NN + nn) * INNER + c] * wr[k];
    }
    z[i] = (__bf16)(acc * (1.f / (1.f + expf(-acc))));
}

// ---------------------------------------------------------------------------
// Host-side launcher
// ---------------------------------------------------------------------------
template <int TA, int TB, int CB>
static inline void launch_gemm(hipStream_t s, const __bf16* A, const __bf16* B,
                               const float* bias, const float* resid, const float* scale,
                               void* C, int M, int N, int K,
                               int lda, int ldb, int ldc,
                               long long sA, long long sB, long long sC,
                               long long sScale, float alpha, int batch)
{
    dim3 grid((N + BN - 1) / BN, (M + BM - 1) / BM, batch);
    wmma_gemm<TA, TB, CB><<<grid, 256, 0, s>>>(A, B, bias, resid, scale, C,
                                               M, N, K, lda, ldb, ldc,
                                               sA, sB, sC, sScale, alpha);
}

extern "C" void kernel_launch(void* const* d_in, const int* in_sizes, int n_in,
                              void* d_out, int out_size, void* d_ws, size_t ws_size,
                              hipStream_t stream)
{
    (void)in_sizes; (void)n_in; (void)out_size; (void)ws_size;

    const float* x_in  = (const float*)d_in[0];
    const float* ln1_g = (const float*)d_in[1];
    const float* ln1_b = (const float*)d_in[2];
    const float* wq    = (const float*)d_in[3];
    const float* bq    = (const float*)d_in[4];
    const float* wk    = (const float*)d_in[5];
    const float* bk    = (const float*)d_in[6];
    const float* wv    = (const float*)d_in[7];
    const float* bv    = (const float*)d_in[8];
    const float* wo    = (const float*)d_in[9];
    const float* bo    = (const float*)d_in[10];
    const float* proj  = (const float*)d_in[11];
    const float* ln2_g = (const float*)d_in[12];
    const float* ln2_b = (const float*)d_in[13];
    const float* pw1_w = (const float*)d_in[14];
    const float* pw1_b = (const float*)d_in[15];
    const float* dw_w  = (const float*)d_in[16];
    const float* dw_b  = (const float*)d_in[17];
    const float* pw2_w = (const float*)d_in[18];
    const float* pw2_b = (const float*)d_in[19];
    float* out = (float*)d_out;

    const size_t R  = (size_t)BB * NN;
    const size_t RD = R * DD;
    const size_t FQ = (size_t)BB * HH * NN * MF;

    char* base = (char*)d_ws;
    size_t off = 0;
    auto alloc = [&](size_t bytes) -> void* {
        void* p = base + off;
        off += (bytes + 255) & ~(size_t)255;
        return p;
    };

    float*  buf_x    = (float*)alloc(RD * 4);
    __bf16* buf_y    = (__bf16*)alloc(RD * 2);
    __bf16* buf_q    = (__bf16*)alloc(RD * 2);     // reused as outh
    __bf16* buf_k    = (__bf16*)alloc(RD * 2);     // reused as merged out
    __bf16* buf_v    = (__bf16*)alloc(RD * 2);
    __bf16* buf_qh   = (__bf16*)alloc(RD * 2);
    __bf16* buf_kh   = (__bf16*)alloc(RD * 2);
    __bf16* buf_vh   = (__bf16*)alloc(RD * 2);
    float*  buf_dd   = (float*)alloc(FQ * 4);      // shared fp32 scratch for q/k features
    __bf16* buf_qkb  = (__bf16*)alloc(FQ * 2);
    __bf16* buf_kkb  = (__bf16*)alloc(FQ * 2);
    __bf16* buf_ctx  = (__bf16*)alloc((size_t)BB * HH * DH * MF * 2);
    float*  buf_ksum = (float*)alloc((size_t)BB * HH * MF * 4);
    float*  buf_dinv = (float*)alloc((size_t)BB * HH * NN * 4);
    float*  buf_pw1  = (float*)alloc(R * (size_t)(2 * INNER) * 4);
    float*  buf_glu  = (float*)alloc(R * (size_t)INNER * 4);
    __bf16* buf_conv = (__bf16*)alloc(R * (size_t)INNER * 2);
    // bf16 weight copies (all layers)
    __bf16* wqb   = (__bf16*)alloc((size_t)LL * DD * DD * 2);
    __bf16* wkb   = (__bf16*)alloc((size_t)LL * DD * DD * 2);
    __bf16* wvb   = (__bf16*)alloc((size_t)LL * DD * DD * 2);
    __bf16* wob   = (__bf16*)alloc((size_t)LL * DD * DD * 2);
    __bf16* projb = (__bf16*)alloc((size_t)LL * MF * DH * 2);
    __bf16* pw1wb = (__bf16*)alloc((size_t)LL * (2 * INNER) * DD * 2);
    __bf16* pw2wb = (__bf16*)alloc((size_t)LL * DD * INNER * 2);

    auto cvt = [&](const float* s, __bf16* d, long long n) {
        cvt_bf16_kernel<<<(int)((n + 255) / 256), 256, 0, stream>>>(s, d, n);
    };
    cvt(wq, wqb, (long long)LL * DD * DD);
    cvt(wk, wkb, (long long)LL * DD * DD);
    cvt(wv, wvb, (long long)LL * DD * DD);
    cvt(wo, wob, (long long)LL * DD * DD);
    cvt(proj, projb, (long long)LL * MF * DH);
    cvt(pw1_w, pw1wb, (long long)LL * (2 * INNER) * DD);
    cvt(pw2_w, pw2wb, (long long)LL * DD * INNER);

    const float normalizer = powf((float)DH, -0.25f);
    const float nh    = 0.5f * normalizer * normalizer;   // 1/16
    const float ratio = rsqrtf((float)MF);

    const int elem_blocks = (int)((RD + 255) / 256);
    const int chan_blocks = (int)((R * INNER + 255) / 256);

    for (int i = 0; i < LL; ++i) {
        const float* x_cur = (i == 0) ? x_in : buf_x;

        // ---- LN1 (writes bf16) ----
        ln_kernel<<<(int)R, 256, 0, stream>>>(x_cur, ln1_g + i * DD, ln1_b + i * DD, buf_y, DD);

        // ---- Q/K/V projections: y @ W^T + b  (bf16 out) ----
        launch_gemm<0, 1, 1>(stream, buf_y, wqb + (size_t)i * DD * DD, bq + i * DD,
                             nullptr, nullptr, buf_q, (int)R, DD, DD, DD, DD, DD,
                             0, 0, 0, 0, 1.f, 1);
        launch_gemm<0, 1, 1>(stream, buf_y, wkb + (size_t)i * DD * DD, bk + i * DD,
                             nullptr, nullptr, buf_k, (int)R, DD, DD, DD, DD, DD,
                             0, 0, 0, 0, 1.f, 1);
        launch_gemm<0, 1, 1>(stream, buf_y, wvb + (size_t)i * DD * DD, bv + i * DD,
                             nullptr, nullptr, buf_v, (int)R, DD, DD, DD, DD, DD,
                             0, 0, 0, 0, 1.f, 1);

        split_heads_kernel<<<elem_blocks, 256, 0, stream>>>(buf_q, buf_qh);
        split_heads_kernel<<<elem_blocks, 256, 0, stream>>>(buf_k, buf_kh);
        split_heads_kernel<<<elem_blocks, 256, 0, stream>>>(buf_v, buf_vh);

        // ---- FAVOR features (batched over B*H), then exp epilogues ----
        const __bf16* pr = projb + (size_t)i * MF * DH;
        launch_gemm<0, 1, 0>(stream, buf_qh, pr, nullptr, nullptr, nullptr, buf_dd,
                             NN, MF, DH, DH, DH, MF,
                             (long long)NN * DH, 0, (long long)NN * MF, 0,
                             normalizer, BB * HH);
        feat_kernel<<<(int)(BB * HH * NN), 128, 0, stream>>>(buf_dd, buf_qh, buf_qkb,
                                                             nh, ratio, 1);
        launch_gemm<0, 1, 0>(stream, buf_kh, pr, nullptr, nullptr, nullptr, buf_dd,
                             NN, MF, DH, DH, DH, MF,
                             (long long)NN * DH, 0, (long long)NN * MF, 0,
                             normalizer, BB * HH);
        feat_kernel<<<(int)(BB * HH * NN), 128, 0, stream>>>(buf_dd, buf_kh, buf_kkb,
                                                             nh, ratio, 0);

        // ---- ksum, d_inv ----
        ksum_kernel<<<BB * HH, 256, 0, stream>>>(buf_kkb, buf_ksum);
        dinv_kernel<<<(BB * HH * NN) / 8, 256, 0, stream>>>(buf_qkb, buf_ksum, buf_dinv);

        // ---- ctx' = v^T @ kk  -> [DH x MF] per (b,h), bf16 ----
        launch_gemm<1, 0, 1>(stream, buf_vh, buf_kkb, nullptr, nullptr, nullptr, buf_ctx,
                             DH, MF, NN, DH, MF, MF,
                             (long long)NN * DH, (long long)NN * MF,
                             (long long)DH * MF, 0, 1.f, BB * HH);

        // ---- outh = d_inv * (qk @ ctx'^T)  -> [N x DH] per (b,h), bf16 ----
        launch_gemm<0, 1, 1>(stream, buf_qkb, buf_ctx, nullptr, nullptr, buf_dinv, buf_q,
                             NN, DH, MF, MF, MF, DH,
                             (long long)NN * MF, (long long)DH * MF,
                             (long long)NN * DH, (long long)NN, 1.f, BB * HH);

        merge_heads_kernel<<<elem_blocks, 256, 0, stream>>>(buf_q, buf_k);

        // ---- x = x + out @ wo^T + bo  (fp32 residual stream) ----
        launch_gemm<0, 1, 0>(stream, buf_k, wob + (size_t)i * DD * DD, bo + i * DD,
                             x_cur, nullptr, buf_x, (int)R, DD, DD, DD, DD, DD,
                             0, 0, 0, 0, 1.f, 1);

        // ---- LN2 ----
        ln_kernel<<<(int)R, 256, 0, stream>>>(buf_x, ln2_g + i * DD, ln2_b + i * DD, buf_y, DD);

        // ---- pw1 ----
        launch_gemm<0, 1, 0>(stream, buf_y, pw1wb + (size_t)i * (2 * INNER) * DD,
                             pw1_b + (size_t)i * (2 * INNER), nullptr, nullptr,
                             buf_pw1, (int)R, 2 * INNER, DD, DD, DD, 2 * INNER,
                             0, 0, 0, 0, 1.f, 1);

        // ---- GLU -> depthwise conv + SiLU (bf16 out) ----
        glu_kernel<<<chan_blocks, 256, 0, stream>>>(buf_pw1, buf_glu);
        dwconv_silu_kernel<<<chan_blocks, 256, 0, stream>>>(
            buf_glu, dw_w + (size_t)i * INNER * KCONV, dw_b + (size_t)i * INNER, buf_conv);

        // ---- pw2 + residual ----
        float* dst = (i == LL - 1) ? out : buf_x;
        launch_gemm<0, 1, 0>(stream, buf_conv, pw2wb + (size_t)i * DD * INNER,
                             pw2_b + (size_t)i * DD, buf_x, nullptr,
                             dst, (int)R, DD, INNER, INNER, INNER, DD,
                             0, 0, 0, 0, 1.f, 1);
    }
}
